// SimpleAttnLayer_20418274525780
// MI455X (gfx1250) — matrix operand order repaired and checked by reading, not compiled, not run on hardware
//
#include <hip/hip_runtime.h>
#include <hip/hip_bf16.h>

#define BB 64
#define NN 8192
#define HH 64
#define AA 128

typedef __attribute__((ext_vector_type(16))) _Float16 v16h;
typedef __attribute__((ext_vector_type(8)))  float    v8f;

// ---------------------------------------------------------------------------
// Fast branch-free tanh. CDNA5 has a hardware V_TANH_F32 transcendental;
// use it when the builtin is exposed, else a 6-op exp2/rcp identity
// (correct at both extremes: exp2->inf gives t=1, sign restored via copysign).
// ---------------------------------------------------------------------------
#if __has_builtin(__builtin_amdgcn_tanhf)
static __device__ __forceinline__ float fast_tanh(float x) {
    return __builtin_amdgcn_tanhf(x);
}
#else
static __device__ __forceinline__ float fast_tanh(float x) {
    const float ax = __builtin_fabsf(x);
    const float e  = __builtin_amdgcn_exp2f(ax * 2.885390081777927f); // 2*log2(e)
    const float t  = 1.0f - 2.0f * __builtin_amdgcn_rcpf(e + 1.0f);
    return __builtin_copysignf(t, x);
}
#endif

// ---------------------------------------------------------------------------
// Kernel 1: scores[b,n] = tanh(x[b,n,:] @ W + bias) . u   via f16 WMMA
//   grid = B*N/128 blocks, 256 threads (8 waves); each wave = 16 rows.
//   W is staged in LDS pre-swizzled into the WMMA B-operand lane layout:
//   fragment f = j*2 + kh (j = A-col tile 0..7, kh = K half 0..1),
//   lane l element e -> W[32*kh + e + 16*(l>>4)][16*j + (l&15)]
// ---------------------------------------------------------------------------
__global__ __launch_bounds__(256) void attn_scores_kernel(
    const float* __restrict__ x, const float* __restrict__ w,
    const float* __restrict__ bias, const float* __restrict__ u,
    float* __restrict__ scores)
{
    __shared__ __align__(32) _Float16 sW[16 * 32 * 16];  // 16 KB
    __shared__ float sBias[AA];
    __shared__ float sU[AA];

    const int tid = threadIdx.x;

    // Cooperative fill of W fragments (8192 elements, 32 per thread)
    for (int idx = tid; idx < 16 * 32 * 16; idx += 256) {
        const int f = idx >> 9;          // fragment index
        const int l = (idx >> 4) & 31;   // lane
        const int e = idx & 15;          // element within lane
        const int j  = f >> 1;
        const int kh = f & 1;
        const int col = (j << 4) + (l & 15);
        const int k   = (kh << 5) + e + ((l >> 4) << 4);
        sW[idx] = (_Float16)w[k * AA + col];
    }
    if (tid < AA) { sBias[tid] = bias[tid]; sU[tid] = u[tid]; }
    __syncthreads();

    const int wave = tid >> 5;
    const int lane = tid & 31;
    const int rowBlock = blockIdx.x * 128 + wave * 16;  // global (b,n) row
    const int b  = rowBlock / NN;
    const int n0 = rowBlock % NN;
    const int m  = lane & 15;     // row within 16-row tile
    const int hi = lane >> 4;     // lane half

    const float* xrow = x + ((size_t)b * NN + (size_t)(n0 + m)) * HH;

    // Build A fragments for K=0..31 (kh=0) and K=32..63 (kh=1).
    // A k order used here (same as B): lane half hi holds K={16hi+0..15} per K-half.
    v16h a[2];
    #pragma unroll
    for (int kh = 0; kh < 2; ++kh) {
        const int kbase = (kh << 5) + (hi << 4);
        #pragma unroll
        for (int c = 0; c < 2; ++c) {           // two contiguous 8-float chunks
            const float4* p = (const float4*)(xrow + kbase + c * 8);
            const float4 f0 = p[0];
            const float4 f1 = p[1];
            const int eb = c * 8;
            a[kh][eb + 0] = (_Float16)f0.x; a[kh][eb + 1] = (_Float16)f0.y;
            a[kh][eb + 2] = (_Float16)f0.z; a[kh][eb + 3] = (_Float16)f0.w;
            a[kh][eb + 4] = (_Float16)f1.x; a[kh][eb + 5] = (_Float16)f1.y;
            a[kh][eb + 6] = (_Float16)f1.z; a[kh][eb + 7] = (_Float16)f1.w;
        }
    }

    float partial[8] = {0.f, 0.f, 0.f, 0.f, 0.f, 0.f, 0.f, 0.f};
    const int colLane = lane & 15;

    #pragma unroll
    for (int j = 0; j < 8; ++j) {
        const v16h b0 = *(const v16h*)&sW[((j * 2 + 0) << 9) + (lane << 4)];
        const v16h b1 = *(const v16h*)&sW[((j * 2 + 1) << 9) + (lane << 4)];
        v8f c = {};
        c = __builtin_amdgcn_wmma_f32_16x16x32_f16(false, a[0], false, b0,
                                                   (short)0, c, false, false);
        c = __builtin_amdgcn_wmma_f32_16x16x32_f16(false, a[1], false, b1,
                                                   (short)0, c, false, false);
        const int col = (j << 4) + colLane;        // A-dim column for this lane
        const float bv = sBias[col];
        const float uv = sU[col];
        #pragma unroll
        for (int r = 0; r < 8; ++r)
            partial[r] += fast_tanh(c[r] + bv) * uv;   // dot with u, per row
    }

    // C layout: lane holds col N=lane&15, rows M = r + 8*hi.
    // Reduce over the 16 columns held by each lane half.
    #pragma unroll
    for (int off = 8; off >= 1; off >>= 1) {
        #pragma unroll
        for (int r = 0; r < 8; ++r)
            partial[r] += __shfl_xor(partial[r], off, 32);
    }
    if ((lane & 15) == 0) {
        float* srow = scores + (size_t)b * NN + n0 + (hi << 3);
        #pragma unroll
        for (int r = 0; r < 8; ++r) srow[r] = partial[r];
    }
}

// ---------------------------------------------------------------------------
// Kernel 2: per-batch softmax stats: stats[b] = {max, sum(exp(s-max))}
// ---------------------------------------------------------------------------
__global__ __launch_bounds__(256) void attn_stats_kernel(
    const float* __restrict__ scores, float* __restrict__ stats)
{
    __shared__ float red[256];
    const int b = blockIdx.x;
    const int tid = threadIdx.x;
    const float* s = scores + (size_t)b * NN;

    float mx = -3.402823466e+38f;
    for (int n = tid; n < NN; n += 256) mx = fmaxf(mx, s[n]);
    red[tid] = mx; __syncthreads();
    for (int o = 128; o >= 1; o >>= 1) {
        if (tid < o) red[tid] = fmaxf(red[tid], red[tid + o]);
        __syncthreads();
    }
    mx = red[0];
    __syncthreads();

    float sum = 0.f;
    for (int n = tid; n < NN; n += 256)
        sum += __builtin_amdgcn_exp2f((s[n] - mx) * 1.4426950408889634f);
    red[tid] = sum; __syncthreads();
    for (int o = 128; o >= 1; o >>= 1) {
        if (tid < o) red[tid] += red[tid + o];
        __syncthreads();
    }
    if (tid == 0) { stats[b * 2 + 0] = mx; stats[b * 2 + 1] = red[0]; }
}

// ---------------------------------------------------------------------------
// Kernel 3: partials[b,chunk,h] = sum_{n in chunk} exp(s[b,n]-max_b)*x[b,n,h]
//   grid = B*32 blocks; h coalesced across 64 lanes, 4-way n parallelism.
//   x re-read here is L2-resident (134 MB < 192 MB L2).
// ---------------------------------------------------------------------------
__global__ __launch_bounds__(256) void attn_partial_kernel(
    const float* __restrict__ x, const float* __restrict__ scores,
    const float* __restrict__ stats, float* __restrict__ partials)
{
    const int b     = blockIdx.x >> 5;
    const int chunk = blockIdx.x & 31;
    const int tid = threadIdx.x;
    const int h = tid & 63;
    const int g = tid >> 6;                       // 0..3
    const float mx = stats[b * 2 + 0];
    const float* xb = x + (size_t)b * NN * HH;
    const float* sb = scores + (size_t)b * NN;
    const int n0 = chunk * (NN / 32);

    float acc = 0.f;
    for (int n = n0 + g; n < n0 + NN / 32; n += 4) {
        const float e = __builtin_amdgcn_exp2f((sb[n] - mx) * 1.4426950408889634f);
        acc += e * xb[(size_t)n * HH + h];
    }
    __shared__ float red[256];
    red[tid] = acc;
    __syncthreads();
    if (tid < 64) {
        const float t = red[tid] + red[tid + 64] + red[tid + 128] + red[tid + 192];
        partials[((size_t)b * 32 + chunk) * HH + h] = t;
    }
}

// ---------------------------------------------------------------------------
// Kernel 4: out[b,h] = (sum_chunk partials) / sumexp_b
// ---------------------------------------------------------------------------
__global__ __launch_bounds__(64) void attn_final_kernel(
    const float* __restrict__ partials, const float* __restrict__ stats,
    float* __restrict__ out)
{
    const int b = blockIdx.x;
    const int h = threadIdx.x;
    float sum = 0.f;
    #pragma unroll 8
    for (int c = 0; c < 32; ++c)
        sum += partials[((size_t)b * 32 + c) * HH + h];
    out[b * HH + h] = sum / stats[b * 2 + 1];
}

// ---------------------------------------------------------------------------
extern "C" void kernel_launch(void* const* d_in, const int* in_sizes, int n_in,
                              void* d_out, int out_size, void* d_ws, size_t ws_size,
                              hipStream_t stream) {
    (void)in_sizes; (void)n_in; (void)out_size; (void)ws_size;
    const float* x    = (const float*)d_in[0];  // (B,N,H)
    const float* w    = (const float*)d_in[1];  // (H,A)
    const float* bias = (const float*)d_in[2];  // (A,)
    const float* u    = (const float*)d_in[3];  // (A,1)
    float* out = (float*)d_out;                 // (B,H)

    char* ws = (char*)d_ws;
    float* scores   = (float*)ws;                                    // B*N f32 = 2 MB
    float* stats    = (float*)(ws + (size_t)BB * NN * 4);            // B*2 f32
    float* partials = (float*)(ws + (size_t)BB * NN * 4 + 1024);     // B*32*H f32 = 512 KB

    attn_scores_kernel <<<BB * NN / 128, 256, 0, stream>>>(x, w, bias, u, scores);
    attn_stats_kernel  <<<BB,            256, 0, stream>>>(scores, stats);
    attn_partial_kernel<<<BB * 32,       256, 0, stream>>>(x, scores, stats, partials);
    attn_final_kernel  <<<BB,             64, 0, stream>>>(partials, stats, out);
}
